// QstEncoderDnc_4346506904354
// MI455X (gfx1250) — matrix operand order repaired and checked by reading, not compiled, not run on hardware
//
#include <hip/hip_runtime.h>
#include <math.h>

#define DEVI __device__ __forceinline__

typedef __bf16 bf16_t;
typedef __attribute__((ext_vector_type(16))) __bf16 bf16x16;
typedef __attribute__((ext_vector_type(8)))  float  f32x8;

// ---------------- problem constants ----------------
constexpr int V_ = 10000, E_ = 300, H_ = 512, M_ = 256, W_ = 64, R_ = 4, O_ = 512;
constexpr int B_ = 128, T_ = 32;
constexpr int NN_IN = 556;      // E + R*W
constexpr int KIN   = 576;      // NN_IN padded to multiple of 32
constexpr int G4H   = 2048;     // 4*H
constexpr int HEADS = 480;      // 471 head cols padded to multiple of 16
constexpr int KOUT  = 768;      // H + R*W (out layer K)
constexpr float CLIPV = 20.f, EPSV = 1e-6f;

// head column offsets inside the packed [16][480] head buffer
constexpr int HC_RK = 0;    // 256 (tanh)
constexpr int HC_RS = 256;  // 4   (softplus)
constexpr int HC_WK = 260;  // 64  (tanh)
constexpr int HC_WS = 324;  // 1   (softplus)
constexpr int HC_EV = 325;  // 64  (sigmoid)
constexpr int HC_WV = 389;  // 64  (tanh)
constexpr int HC_FG = 453;  // 4   (sigmoid)
constexpr int HC_AG = 457;  // 1   (sigmoid)
constexpr int HC_WG = 458;  // 1   (sigmoid)
constexpr int HC_RM = 459;  // 12  (softmax per triple)

// ---------------- workspace layout (bytes) ----------------
constexpr size_t WS_XEMB = 0;                                     // bf16 [T][B][E]
constexpr size_t WS_W0IH = WS_XEMB + (size_t)T_*B_*E_*2;          // bf16 [2048][576]
constexpr size_t WS_W0HH = WS_W0IH + (size_t)G4H*KIN*2;           // bf16 [2048][512]
constexpr size_t WS_W1IH = WS_W0HH + (size_t)G4H*H_*2;
constexpr size_t WS_W1HH = WS_W1IH + (size_t)G4H*H_*2;
constexpr size_t WS_WHEAD= WS_W1HH + (size_t)G4H*H_*2;            // bf16 [480][512]
constexpr size_t WS_WOUT = WS_WHEAD+ (size_t)HEADS*H_*2;          // bf16 [512][768]
constexpr size_t WS_B0   = WS_WOUT + (size_t)O_*KOUT*2;           // f32 [2048]
constexpr size_t WS_B1   = WS_B0   + (size_t)G4H*4;               // f32 [2048]
constexpr size_t WS_HB   = WS_B1   + (size_t)G4H*4;               // f32 [480]
constexpr size_t WS_OB   = WS_HB   + (size_t)HEADS*4;             // f32 [512]
constexpr size_t WS_LINK = WS_OB   + (size_t)O_*4;                // f32 [B][256][256]
constexpr size_t WS_MEM  = WS_LINK + (size_t)B_*M_*M_*4;          // f32 [B][256][64]

// ---------------- LDS layout (bytes, dynamic) ----------------
constexpr int OFF_IN    = 0;                 // bf16 [16][768]  (xt|rflat / xi|rflat)
constexpr int OFF_H0    = OFF_IN   + 24576;  // bf16 [2][16][512]
constexpr int OFF_H1    = OFF_H0   + 32768;  // bf16 [2][16][512]
constexpr int OFF_HEADS = OFF_H1   + 32768;  // f32  [16][480]
constexpr int OFF_RFLAT = OFF_HEADS+ 30720;  // f32  [16][256]
constexpr int OFF_USAGE = OFF_RFLAT+ 16384;  // f32  [16][256]
constexpr int OFF_WW    = OFF_USAGE+ 16384;  // f32  [16][256]
constexpr int OFF_PREC  = OFF_WW   + 16384;  // f32  [16][256]
constexpr int OFF_CW    = OFF_PREC + 16384;  // f32  [16][4][256] (aliases wcw/alloc early)
constexpr int OFF_RW    = OFF_CW   + 65536;  // f32  [16][4][256]
constexpr int SMEM_BYTES= OFF_RW   + 65536;  // 317440 B  (<= 320KB/WGP)

// ---------------- param pointers ----------------
struct P {
  const int*   q;    const float* emb;
  const float* w0ih; const float* w0hh; const float* b0ih; const float* b0hh;
  const float* w1ih; const float* w1hh; const float* b1ih; const float* b1hh;
  const float* rkw;  const float* rkb;  const float* rsw;  const float* rsb;
  const float* wkw;  const float* wkb;  const float* wsw;  const float* wsb;
  const float* evw;  const float* evb;  const float* wvw;  const float* wvb;
  const float* fgw;  const float* fgb;  const float* agw;  const float* agb;
  const float* wgw;  const float* wgb;  const float* rmw;  const float* rmb;
  const float* outw; const float* outb;
};

// ---------------- small math helpers ----------------
DEVI float sigmoidf_(float x){ return 1.f/(1.f+__expf(-x)); }
DEVI float softplusf_(float x){ return (x > 20.f) ? x : log1pf(__expf(x)); }
DEVI float waveRedSum(float v){
  #pragma unroll
  for(int m=16;m>=1;m>>=1) v += __shfl_xor(v, m, 32);
  return v;
}
DEVI float waveRedMax(float v){
  #pragma unroll
  for(int m=16;m>=1;m>>=1) v = fmaxf(v, __shfl_xor(v, m, 32));
  return v;
}

// ---------------- WMMA fragment helpers ----------------
// A: 16x32 bf16, row = lane%16; lane<16 -> K {kc..kc+7, kc+16..kc+23}; lane>=16 -> +8
DEVI bf16x16 loadA(const bf16_t* base, int ld, int kc, int lane){
  const int m = lane & 15;
  const bf16_t* p = base + (size_t)m*ld + kc + ((lane & 16) ? 8 : 0);
  bf16x16 a;
  #pragma unroll
  for(int q=0;q<8;++q){ a[q] = p[q]; a[8+q] = p[16+q]; }
  return a;
}
// B: 32x16 bf16 from row-major weights [n][K]; col = lane%16; lane<16 -> K {kc..kc+7, kc+16..kc+23}; lane>=16 -> +8
DEVI bf16x16 loadB(const bf16_t* wrow0, int K, int kc, int lane){
  const int n = lane & 15;
  const bf16_t* p = wrow0 + (size_t)n*K + kc + ((lane & 16) ? 8 : 0);
  bf16x16 b;
  #pragma unroll
  for(int q=0;q<8;++q){ b[q] = p[q]; b[8+q] = p[16+q]; }
  return b;
}
DEVI f32x8 wmma_bf16(bf16x16 a, bf16x16 b, f32x8 c){
  return __builtin_amdgcn_wmma_f32_16x16x32_bf16(false, a, false, b, (short)0, c, false, false);
}

// ---------------- fused LSTM cell (one wave = 2 hidden chunks of 16, x 4 gates) ----------------
DEVI void lstm_tile(const bf16_t* A, int lda, int Kih, const bf16_t* hprev,
                    const bf16_t* Wih, const bf16_t* Whh, const float* bias,
                    float* cc, bf16_t* hnext, bf16_t* xiDst, bool doClip,
                    int wave, int lane)
{
  #pragma unroll 1
  for(int half=0; half<2; ++half){
    const int hc = wave + half*16;          // hidden chunk 0..31
    f32x8 acc[4];
    #pragma unroll
    for(int g=0; g<4; ++g)
      #pragma unroll
      for(int r=0; r<8; ++r) acc[g][r] = 0.f;

    for(int kc=0; kc<Kih; kc+=32){
      bf16x16 a = loadA(A, lda, kc, lane);
      #pragma unroll
      for(int g=0; g<4; ++g){
        bf16x16 b = loadB(Wih + (size_t)(g*H_ + hc*16)*Kih, Kih, kc, lane);
        acc[g] = wmma_bf16(a, b, acc[g]);
      }
    }
    for(int kc=0; kc<H_; kc+=32){
      bf16x16 a = loadA(hprev, H_, kc, lane);
      #pragma unroll
      for(int g=0; g<4; ++g){
        bf16x16 b = loadB(Whh + (size_t)(g*H_ + hc*16)*H_, H_, kc, lane);
        acc[g] = wmma_bf16(a, b, acc[g]);
      }
    }
    const int n = lane & 15, col = hc*16 + n, mb = (lane & 16) ? 8 : 0;
    const float bI = bias[col], bF = bias[H_+col], bG = bias[2*H_+col], bO = bias[3*H_+col];
    #pragma unroll
    for(int r=0; r<8; ++r){
      float ig = sigmoidf_(acc[0][r] + bI);
      float fg = sigmoidf_(acc[1][r] + bF);
      float gg = tanhf   (acc[2][r] + bG);
      float og = sigmoidf_(acc[3][r] + bO);
      float c  = fg*cc[half*8+r] + ig*gg;
      cc[half*8+r] = c;
      float h  = og*tanhf(c);
      const int m = mb + r;
      hnext[(size_t)m*H_ + col] = (bf16_t)h;
      if(doClip){
        float xi = fminf(CLIPV, fmaxf(-CLIPV, h));
        xiDst[(size_t)m*KOUT + col] = (bf16_t)xi;
      }
    }
  }
}

// =================== prep kernels ===================
__global__ void prep_emb(P p, char* ws){
  bf16_t* xe = (bf16_t*)(ws + WS_XEMB);
  const int total = T_*B_*E_;
  for(int idx = blockIdx.x*blockDim.x + threadIdx.x; idx < total; idx += gridDim.x*blockDim.x){
    int e = idx % E_; int bt = idx / E_; int b = bt % B_; int t = bt / B_;
    int q = p.q[b*T_ + t];
    xe[idx] = (bf16_t)tanhf(p.emb[(size_t)q*E_ + e]);
  }
}

__global__ void prep_cvt(const float* src, bf16_t* dst, int rows, int scols, int dcols){
  const long total = (long)rows*dcols;
  for(long idx = blockIdx.x*(long)blockDim.x + threadIdx.x; idx < total; idx += gridDim.x*(long)blockDim.x){
    int r = (int)(idx / dcols), c = (int)(idx % dcols);
    dst[idx] = (c < scols) ? (bf16_t)src[(size_t)r*scols + c] : (bf16_t)0.f;
  }
}

DEVI const float* head_src(int r, const P& p, int* sr, const float** bsrc){
  if(r < 256){ *sr = r;     *bsrc = p.rkb; return p.rkw; }
  if(r < 260){ *sr = r-256; *bsrc = p.rsb; return p.rsw; }
  if(r < 324){ *sr = r-260; *bsrc = p.wkb; return p.wkw; }
  if(r ==324){ *sr = 0;     *bsrc = p.wsb; return p.wsw; }
  if(r < 389){ *sr = r-325; *bsrc = p.evb; return p.evw; }
  if(r < 453){ *sr = r-389; *bsrc = p.wvb; return p.wvw; }
  if(r < 457){ *sr = r-453; *bsrc = p.fgb; return p.fgw; }
  if(r ==457){ *sr = 0;     *bsrc = p.agb; return p.agw; }
  if(r ==458){ *sr = 0;     *bsrc = p.wgb; return p.wgw; }
  if(r < 471){ *sr = r-459; *bsrc = p.rmb; return p.rmw; }
  *sr = 0; *bsrc = nullptr; return nullptr;
}

__global__ void prep_head(P p, char* ws){
  bf16_t* wh = (bf16_t*)(ws + WS_WHEAD);
  float*  hb = (float*) (ws + WS_HB);
  const int total = HEADS*H_;
  for(int idx = blockIdx.x*blockDim.x + threadIdx.x; idx < total + HEADS; idx += gridDim.x*blockDim.x){
    if(idx < total){
      int r = idx / H_, k = idx % H_, sr; const float* bsrc;
      const float* sw = head_src(r, p, &sr, &bsrc);
      wh[idx] = sw ? (bf16_t)sw[(size_t)sr*H_ + k] : (bf16_t)0.f;
    } else {
      int r = idx - total, sr; const float* bsrc;
      head_src(r, p, &sr, &bsrc);
      hb[r] = bsrc ? bsrc[sr] : 0.f;
    }
  }
}

__global__ void prep_bias(P p, char* ws){
  float* b0 = (float*)(ws + WS_B0);
  float* b1 = (float*)(ws + WS_B1);
  float* ob = (float*)(ws + WS_OB);
  const int total = 2*G4H + O_;
  for(int idx = blockIdx.x*blockDim.x + threadIdx.x; idx < total; idx += gridDim.x*blockDim.x){
    if(idx < G4H)            b0[idx]        = p.b0ih[idx]       + p.b0hh[idx];
    else if(idx < 2*G4H)     b1[idx-G4H]    = p.b1ih[idx-G4H]   + p.b1hh[idx-G4H];
    else                     ob[idx-2*G4H]  = p.outb[idx-2*G4H];
  }
}

// =================== persistent DNC kernel: 8 blocks x 512 threads (16 waves) ===================
__global__ __launch_bounds__(512, 1)
void dnc_main(char* __restrict__ ws, float* __restrict__ out)
{
  extern __shared__ char smem[];
  bf16_t* sIn    = (bf16_t*)(smem + OFF_IN);     // [16][768]
  bf16_t* sH0    = (bf16_t*)(smem + OFF_H0);     // [2][16][512]
  bf16_t* sH1    = (bf16_t*)(smem + OFF_H1);     // [2][16][512]
  float*  sHeads = (float*) (smem + OFF_HEADS);  // [16][480]
  float*  sRflat = (float*) (smem + OFF_RFLAT);  // [16][256]
  float*  sUsage = (float*) (smem + OFF_USAGE);  // [16][256]
  float*  sWW    = (float*) (smem + OFF_WW);     // [16][256]
  float*  sPrec  = (float*) (smem + OFF_PREC);   // [16][256]
  float*  sCW    = (float*) (smem + OFF_CW);     // [16][4][256]
  float*  sWcw   = sCW;                          // alias (dead before cw is built)
  float*  sAlloc = sCW + 16*256;                 // alias
  float*  sRW    = (float*) (smem + OFF_RW);     // [16][4][256]

  const bf16_t* xemb  = (const bf16_t*)(ws + WS_XEMB);
  const bf16_t* w0ih  = (const bf16_t*)(ws + WS_W0IH);
  const bf16_t* w0hh  = (const bf16_t*)(ws + WS_W0HH);
  const bf16_t* w1ih  = (const bf16_t*)(ws + WS_W1IH);
  const bf16_t* w1hh  = (const bf16_t*)(ws + WS_W1HH);
  const bf16_t* whead = (const bf16_t*)(ws + WS_WHEAD);
  const bf16_t* wout  = (const bf16_t*)(ws + WS_WOUT);
  const float*  b0    = (const float*) (ws + WS_B0);
  const float*  b1    = (const float*) (ws + WS_B1);
  const float*  hbias = (const float*) (ws + WS_HB);
  const float*  obias = (const float*) (ws + WS_OB);
  float* linkG = (float*)(ws + WS_LINK);
  float* memG  = (float*)(ws + WS_MEM);

  const int tid = threadIdx.x, wave = tid >> 5, lane = tid & 31;
  const int grp = blockIdx.x, bg0 = grp*16, bi = wave;

  // ---- init recurrent state (per launch, deterministic) ----
  for(int i=tid;i<16*KOUT;i+=512) sIn[i] = (bf16_t)0.f;
  for(int i=tid;i<2*16*H_;i+=512){ sH0[i] = (bf16_t)0.f; sH1[i] = (bf16_t)0.f; }
  for(int i=tid;i<16*M_;i+=512){ sRflat[i]=0.f; sUsage[i]=0.f; sWW[i]=EPSV; sPrec[i]=0.f; }
  for(int i=tid;i<16*R_*M_;i+=512) sRW[i] = EPSV;
  for(size_t i=tid;i<(size_t)16*M_*M_;i+=512) linkG[(size_t)bg0*M_*M_ + i] = 0.f;
  for(int i=tid;i<16*M_*W_;i+=512) memG[(size_t)bg0*M_*W_ + i] = EPSV;

  float c0[16], c1[16];
  #pragma unroll
  for(int i=0;i<16;++i){ c0[i]=0.f; c1[i]=0.f; }
  f32x8 yacc[2];
  #pragma unroll
  for(int pss=0;pss<2;++pss)
    #pragma unroll
    for(int r=0;r<8;++r) yacc[pss][r]=0.f;
  __syncthreads();

  for(int t=0; t<T_; ++t){
    const int pcur = t & 1, pnxt = (t + 1) & 1;
    bf16_t* h0prev = sH0 + pcur*16*H_;  bf16_t* h0next = sH0 + pnxt*16*H_;
    bf16_t* h1prev = sH1 + pcur*16*H_;  bf16_t* h1next = sH1 + pnxt*16*H_;

    // ---- phase 1: build lstm0 input [xt | rflat | pad] (bf16) ----
    for(int idx=tid; idx<16*KIN; idx+=512){
      int b = idx / KIN, col = idx % KIN;
      bf16_t v;
      if(col < E_)          v = xemb[((size_t)t*B_ + bg0 + b)*E_ + col];
      else if(col < NN_IN)  v = (bf16_t)sRflat[b*M_ + (col - E_)];
      else                  v = (bf16_t)0.f;
      sIn[(size_t)b*KOUT + col] = v;
    }
    __syncthreads();

    // ---- phase 2/3: LSTM0 then LSTM1 (bf16 WMMA) ----
    lstm_tile(sIn, KOUT, KIN, h0prev, w0ih, w0hh, b0, c0, h0next, nullptr, false, wave, lane);
    __syncthreads();
    lstm_tile(h0next, H_, H_, h1prev, w1ih, w1hh, b1, c1, h1next, sIn, true, wave, lane);
    __syncthreads();

    // ---- phase 4: all heads as one [16,480] = [16,512]x[512,480] GEMM ----
    #pragma unroll 1
    for(int pss=0; pss<2; ++pss){
      int tile = wave + pss*16;
      if(tile < HEADS/16){
        f32x8 acc;
        #pragma unroll
        for(int r=0;r<8;++r) acc[r]=0.f;
        for(int kc=0; kc<H_; kc+=32){
          bf16x16 a = loadA(sIn, KOUT, kc, lane);
          bf16x16 b = loadB(whead + (size_t)tile*16*H_, H_, kc, lane);
          acc = wmma_bf16(a, b, acc);
        }
        int n = lane&15, col = tile*16+n, mb = (lane&16)?8:0;
        float hb = hbias[col];
        #pragma unroll
        for(int r=0;r<8;++r) sHeads[(size_t)(mb+r)*HEADS + col] = acc[r] + hb;
      }
    }
    __syncthreads();

    // ---- phase 4.5: in-place activations on head outputs ----
    for(int idx=tid; idx<16*HEADS; idx+=512){
      int col = idx % HEADS;
      float* pp = &sHeads[idx]; float v = *pp;
      if(col < HC_RS)        *pp = tanhf(v);
      else if(col < HC_WK)   *pp = softplusf_(v);
      else if(col < HC_WS)   *pp = tanhf(v);
      else if(col == HC_WS)  *pp = softplusf_(v);
      else if(col < HC_WV)   *pp = sigmoidf_(v);
      else if(col < HC_FG)   *pp = tanhf(v);
      else if(col < HC_RM)   *pp = sigmoidf_(v);       // fg, ag, wg
      else if(col < HC_RM+12){
        if(((col - HC_RM) % 3) == 0){                  // softmax over triple (read modes)
          float a = v, bb = pp[1], cc = pp[2];
          float mx = fmaxf(a, fmaxf(bb, cc));
          float ea = __expf(a-mx), eb = __expf(bb-mx), ec = __expf(cc-mx);
          float s = ea+eb+ec; pp[0]=ea/s; pp[1]=eb/s; pp[2]=ec/s;
        }
      }
    }
    __syncthreads();

    // ---- phase 5a: usage update (uses prev ww, prev rw) ----
    for(int idx=tid; idx<16*M_; idx+=512){
      int b = idx >> 8, m = idx & 255;
      float u = sUsage[idx], wwp = sWW[idx];
      u = u + (1.f - u)*wwp;
      const float* hb_ = &sHeads[b*HEADS];
      float psi = 1.f;
      #pragma unroll
      for(int r=0;r<R_;++r) psi *= (1.f - hb_[HC_FG+r]*sRW[(b*R_+r)*M_ + m]);
      sUsage[idx] = u*psi;
    }
    __syncthreads();

    // ---- phase 5b/5c (wave = one batch element): write content weights + allocation ----
    {
      const float* hb_ = &sHeads[bi*HEADS];
      float wstr = hb_[HC_WS];
      float kn2 = 0.f;
      for(int j=0;j<W_;++j){ float k = hb_[HC_WK+j]; kn2 += k*k; }
      float knorm = sqrtf(kn2) + EPSV;
      float logit[8]; float lmax = -1e30f;
      #pragma unroll 1
      for(int c=0;c<8;++c){
        int m = lane + 32*c;
        const float* mrow = &memG[((size_t)(bg0+bi)*M_ + m)*W_];
        float dot = 0.f, n2 = 0.f;
        for(int j=0;j<W_;++j){ float mv = mrow[j]; dot += mv*hb_[HC_WK+j]; n2 += mv*mv; }
        float s = dot / ((sqrtf(n2)+EPSV)*knorm) * wstr;
        logit[c] = s; lmax = fmaxf(lmax, s);
      }
      lmax = waveRedMax(lmax);
      float lsum = 0.f;
      #pragma unroll
      for(int c=0;c<8;++c){ float e = __expf(logit[c]-lmax); logit[c]=e; lsum+=e; }
      lsum = waveRedSum(lsum);
      #pragma unroll
      for(int c=0;c<8;++c) sWcw[bi*M_ + lane + 32*c] = logit[c]/lsum;

      // allocation: stable-ascending cumprod product (equivalent to argsort path)
      #pragma unroll 1
      for(int c=0;c<8;++c){
        int m = lane + 32*c;
        float um = EPSV + (1.f-EPSV)*sUsage[bi*M_ + m];
        float prod = 1.f;
        for(int j=0;j<M_;++j){
          float uj = EPSV + (1.f-EPSV)*sUsage[bi*M_ + j];
          bool before = (uj < um) || (uj == um && j < m);
          prod *= before ? uj : 1.f;
        }
        sAlloc[bi*M_ + m] = (1.f - um)*prod;
      }
    }
    __syncthreads();

    // ---- phase 5d: write weighting ----
    for(int idx=tid; idx<16*M_; idx+=512){
      int b = idx >> 8;
      const float* hb_ = &sHeads[b*HEADS];
      float ag = hb_[HC_AG], wg = hb_[HC_WG];
      sWW[idx] = wg*(ag*sAlloc[idx] + (1.f-ag)*sWcw[idx]);
    }
    __syncthreads();

    // ---- phase 5e: memory erase + write (global RMW, L2-resident) ----
    for(int idx=tid; idx<16*M_*W_; idx+=512){
      int b = idx/(M_*W_), rem = idx%(M_*W_), m = rem/W_, w = rem%W_;
      float wwv = sWW[b*M_ + m];
      const float* hb_ = &sHeads[b*HEADS];
      float* pm = &memG[(size_t)(bg0+b)*M_*W_ + rem];
      *pm = *pm * (1.f - wwv*hb_[HC_EV+w]) + wwv*hb_[HC_WV+w];
    }
    __syncthreads();

    // ---- phase 5f/5g/5h (wave = one batch): link RMW + fused fwd/bwd, read content, rw update ----
    {
      const float* hb_ = &sHeads[bi*HEADS];
      float rwold[R_][8], precj[8], wwj[8], bwd[R_][8];
      float wwpart = 0.f;
      #pragma unroll
      for(int c=0;c<8;++c){
        int j = lane + 32*c;
        wwj[c]   = sWW[bi*M_ + j];
        precj[c] = sPrec[bi*M_ + j];
        wwpart  += wwj[c];
        #pragma unroll
        for(int r=0;r<R_;++r){ rwold[r][c] = sRW[(bi*R_+r)*M_ + j]; bwd[r][c] = 0.f; }
      }
      float wwsum = waveRedSum(wwpart);
      float* Lbase = &linkG[(size_t)(bg0+bi)*M_*M_];

      #pragma unroll 1
      for(int i=0;i<M_;++i){
        float wwi = sWW[bi*M_ + i];
        float rwi[R_];
        #pragma unroll
        for(int r=0;r<R_;++r) rwi[r] = sRW[(bi*R_+r)*M_ + i];   // still old at iter i
        float fpart[R_] = {0.f,0.f,0.f,0.f};
        float* Lrow = Lbase + (size_t)i*M_;
        #pragma unroll
        for(int c=0;c<8;++c){
          int j = lane + 32*c;
          float L  = Lrow[j];
          float nl = (1.f - wwi - wwj[c])*L + wwi*precj[c];
          if(j == i) nl = 0.f;
          Lrow[j] = nl;
          #pragma unroll
          for(int r=0;r<R_;++r){ fpart[r] += nl*rwold[r][c]; bwd[r][c] += rwi[r]*nl; }
        }
        #pragma unroll
        for(int r=0;r<R_;++r){
          float f = waveRedSum(fpart[r]);
          if(lane == 0) sRW[(bi*R_+r)*M_ + i] = f;              // overwrite rw with fwd
        }
      }
      // precedence update (after link pass)
      #pragma unroll
      for(int c=0;c<8;++c){
        int j = lane + 32*c;
        sPrec[bi*M_ + j] = (1.f - wwsum)*precj[c] + wwj[c];
      }

      // read content weights (over updated memory)
      float knormR[R_], rstr[R_];
      #pragma unroll 1
      for(int r=0;r<R_;++r){
        float kn2 = 0.f;
        for(int j=0;j<W_;++j){ float k = hb_[HC_RK + r*W_ + j]; kn2 += k*k; }
        knormR[r] = sqrtf(kn2) + EPSV;
        rstr[r]   = hb_[HC_RS + r];
      }
      #pragma unroll 1
      for(int c=0;c<8;++c){
        int m = lane + 32*c;
        const float* mrow = &memG[((size_t)(bg0+bi)*M_ + m)*W_];
        float dot[R_] = {0.f,0.f,0.f,0.f}; float n2 = 0.f;
        for(int j=0;j<W_;++j){
          float mv = mrow[j]; n2 += mv*mv;
          #pragma unroll
          for(int r=0;r<R_;++r) dot[r] += mv*hb_[HC_RK + r*W_ + j];
        }
        float invm = 1.f/(sqrtf(n2)+EPSV);
        #pragma unroll
        for(int r=0;r<R_;++r) sCW[(bi*R_+r)*M_ + m] = dot[r]*invm/knormR[r]*rstr[r];
      }
      #pragma unroll 1
      for(int r=0;r<R_;++r){
        float lm = -1e30f;
        #pragma unroll
        for(int c=0;c<8;++c) lm = fmaxf(lm, sCW[(bi*R_+r)*M_ + lane + 32*c]);
        lm = waveRedMax(lm);
        float ls = 0.f;
        #pragma unroll
        for(int c=0;c<8;++c){
          float e = __expf(sCW[(bi*R_+r)*M_ + lane + 32*c] - lm);
          sCW[(bi*R_+r)*M_ + lane + 32*c] = e; ls += e;
        }
        ls = waveRedSum(ls);
        float inv = 1.f/ls;
        #pragma unroll
        for(int c=0;c<8;++c) sCW[(bi*R_+r)*M_ + lane + 32*c] *= inv;
      }
      // rw = m0*bwd + m1*fwd + m2*cw
      #pragma unroll
      for(int r=0;r<R_;++r){
        float m0 = hb_[HC_RM + r*3 + 0], m1 = hb_[HC_RM + r*3 + 1], m2 = hb_[HC_RM + r*3 + 2];
        #pragma unroll
        for(int c=0;c<8;++c){
          int j = lane + 32*c;
          float fwd = sRW[(bi*R_+r)*M_ + j];
          sRW[(bi*R_+r)*M_ + j] = m0*bwd[r][c] + m1*fwd + m2*sCW[(bi*R_+r)*M_ + j];
        }
      }
    }
    __syncthreads();

    // ---- rvec = rw @ mem ; build rflat (f32 + bf16 copy for the out layer) ----
    for(int idx=tid; idx<16*R_*W_; idx+=512){
      int b = idx/(R_*W_), rwi = idx%(R_*W_), r = rwi/W_, w = rwi%W_;
      const float* mb_ = &memG[(size_t)(bg0+b)*M_*W_ + w];
      const float* rwv = &sRW[(b*R_+r)*M_];
      float acc = 0.f;
      for(int m=0;m<M_;++m) acc += rwv[m]*mb_[(size_t)m*W_];
      sRflat[b*M_ + rwi] = acc;
      sIn[(size_t)b*KOUT + H_ + rwi] = (bf16_t)acc;
    }
    __syncthreads();

    // ---- phase 6: output layer, accumulated over t in registers ----
    #pragma unroll 1
    for(int pss=0; pss<2; ++pss){
      int tile = wave + pss*16;
      for(int kc=0; kc<KOUT; kc+=32){
        bf16x16 a = loadA(sIn, KOUT, kc, lane);
        bf16x16 b = loadB(wout + (size_t)tile*16*KOUT, KOUT, kc, lane);
        yacc[pss] = wmma_bf16(a, b, yacc[pss]);
      }
    }
    __syncthreads();
  }

  // ---- final: mean over T + bias ----
  #pragma unroll
  for(int pss=0; pss<2; ++pss){
    int tile = wave + pss*16, n = lane&15, col = tile*16+n, mb = (lane&16)?8:0;
    float ob = obias[col];
    #pragma unroll
    for(int r=0;r<8;++r){
      int b = mb + r;
      out[(size_t)(bg0+b)*O_ + col] = yacc[pss][r]*(1.f/(float)T_) + ob;
    }
  }
}

// =================== host entry ===================
extern "C" void kernel_launch(void* const* d_in, const int* in_sizes, int n_in,
                              void* d_out, int out_size, void* d_ws, size_t ws_size,
                              hipStream_t stream)
{
  (void)in_sizes; (void)n_in; (void)out_size; (void)ws_size;
  P p;
  p.q    = (const int*)  d_in[0];  p.emb  = (const float*)d_in[1];
  p.w0ih = (const float*)d_in[2];  p.w0hh = (const float*)d_in[3];
  p.b0ih = (const float*)d_in[4];  p.b0hh = (const float*)d_in[5];
  p.w1ih = (const float*)d_in[6];  p.w1hh = (const float*)d_in[7];
  p.b1ih = (const float*)d_in[8];  p.b1hh = (const float*)d_in[9];
  p.rkw  = (const float*)d_in[10]; p.rkb  = (const float*)d_in[11];
  p.rsw  = (const float*)d_in[12]; p.rsb  = (const float*)d_in[13];
  p.wkw  = (const float*)d_in[14]; p.wkb  = (const float*)d_in[15];
  p.wsw  = (const float*)d_in[16]; p.wsb  = (const float*)d_in[17];
  p.evw  = (const float*)d_in[18]; p.evb  = (const float*)d_in[19];
  p.wvw  = (const float*)d_in[20]; p.wvb  = (const float*)d_in[21];
  p.fgw  = (const float*)d_in[22]; p.fgb  = (const float*)d_in[23];
  p.agw  = (const float*)d_in[24]; p.agb  = (const float*)d_in[25];
  p.wgw  = (const float*)d_in[26]; p.wgb  = (const float*)d_in[27];
  p.rmw  = (const float*)d_in[28]; p.rmb  = (const float*)d_in[29];
  p.outw = (const float*)d_in[30]; p.outb = (const float*)d_in[31];

  char* ws = (char*)d_ws;

  prep_emb <<<512, 256, 0, stream>>>(p, ws);
  prep_cvt <<<512, 256, 0, stream>>>(p.w0ih, (bf16_t*)(ws + WS_W0IH), G4H, NN_IN, KIN);
  prep_cvt <<<512, 256, 0, stream>>>(p.w0hh, (bf16_t*)(ws + WS_W0HH), G4H, H_, H_);
  prep_cvt <<<512, 256, 0, stream>>>(p.w1ih, (bf16_t*)(ws + WS_W1IH), G4H, H_, H_);
  prep_cvt <<<512, 256, 0, stream>>>(p.w1hh, (bf16_t*)(ws + WS_W1HH), G4H, H_, H_);
  prep_cvt <<<512, 256, 0, stream>>>(p.outw, (bf16_t*)(ws + WS_WOUT), O_, KOUT, KOUT);
  prep_head<<<256, 256, 0, stream>>>(p, ws);
  prep_bias<<<32,  256, 0, stream>>>(p, ws);

  dnc_main<<<dim3(B_/16), dim3(512), SMEM_BYTES, stream>>>(ws, (float*)d_out);
}
